// Wrapper_10368051052625
// MI455X (gfx1250) — compile-verified, matrix operand order checked by reading, not run on hardware
//
#include <hip/hip_runtime.h>
#include <math.h>

typedef __attribute__((ext_vector_type(16))) _Float16 v16h;
typedef __attribute__((ext_vector_type(8)))  float    v8f;

#define NBOX   8732
#define NCLS   81      // incl background
#define NFG    80
#define TOPK   200
#define MAXOUT 20
#define OH     300
#define OW     300
#define SRC    4096

__constant__ int c_coco[81] = {
  1,2,3,4,5,6,7,8,9,10,11,13,14,15,16,17,18,19,20,21,22,23,24,25,27,28,31,32,
  33,34,35,36,37,38,39,40,41,42,43,44,46,47,48,49,50,51,52,53,54,55,56,57,58,
  59,60,61,62,63,64,65,67,70,72,73,74,75,76,77,78,79,80,81,82,84,85,86,87,88,
  89,90,91};

// ---------------------------------------------------------------------------
// 1) Fused normalize + bilinear resize (align_corners). Gather-only: reads
//    4 pixels per output instead of streaming the whole 201 MB image.
// ---------------------------------------------------------------------------
__global__ __launch_bounds__(256) void resize_kernel(const float* __restrict__ img,
                                                     float* __restrict__ out) {
  int idx = blockIdx.x * blockDim.x + threadIdx.x;
  if (idx >= 3 * OH * OW) return;
  int ox = idx % OW;
  int t  = idx / OW;
  int oy = t % OH;
  int ch = t / OH;
  const float r = (float)(SRC - 1) / (float)(OH - 1);
  float sy = oy * r;
  float sx = ox * r;
  int y0 = (int)floorf(sy); y0 = min(max(y0, 0), SRC - 2);
  int x0 = (int)floorf(sx); x0 = min(max(x0, 0), SRC - 2);
  float wy = sy - (float)y0;
  float wx = sx - (float)x0;
  const float* p = img + (size_t)ch * SRC * SRC;
  float v00 = p[(size_t)y0 * SRC + x0];
  float v01 = p[(size_t)y0 * SRC + x0 + 1];
  float v10 = p[(size_t)(y0 + 1) * SRC + x0];
  float v11 = p[(size_t)(y0 + 1) * SRC + x0 + 1];
  float top = v00 * (1.0f - wx) + v01 * wx;
  float bot = v10 * (1.0f - wx) + v11 * wx;
  float v   = top * (1.0f - wy) + bot * wy;
  out[idx] = (v * 255.0f - 128.0f) * (1.0f / 128.0f);   // normalize (linear, commutes)
}

// ---------------------------------------------------------------------------
// 2) Box decode: xywh deltas -> ltrb
// ---------------------------------------------------------------------------
__global__ __launch_bounds__(256) void decode_kernel(const float* __restrict__ ploc,
                                                     const float* __restrict__ dboxes,
                                                     float* __restrict__ ltrb) {
  int n = blockIdx.x * blockDim.x + threadIdx.x;
  if (n >= NBOX) return;
  float lx = ploc[0 * NBOX + n];
  float ly = ploc[1 * NBOX + n];
  float lw = ploc[2 * NBOX + n];
  float lh = ploc[3 * NBOX + n];
  float dx = dboxes[n * 4 + 0], dy = dboxes[n * 4 + 1];
  float dw = dboxes[n * 4 + 2], dh = dboxes[n * 4 + 3];
  float cx = lx * 0.1f * dw + dx;
  float cy = ly * 0.1f * dh + dy;
  float w  = __expf(lw * 0.2f) * dw;
  float h  = __expf(lh * 0.2f) * dh;
  ltrb[n * 4 + 0] = cx - 0.5f * w;
  ltrb[n * 4 + 1] = cy - 0.5f * h;
  ltrb[n * 4 + 2] = cx + 0.5f * w;
  ltrb[n * 4 + 3] = cy + 0.5f * h;
}

// ---------------------------------------------------------------------------
// 3) Softmax with WMMA row-reduction.
//    denom(row m) = sum_k exp(logit[m,k]) = [exp A](16x32) x ones(32x16).
//    One wave32 per 16 boxes, 3 K-chunks of 32 (classes 0..95, zero padded).
//    A-matrix lane layout (16-bit A, 16x32):
//      lanes 0-15 (M=lane):    half0 -> K c0+0..7,  half1 -> K c0+16..23
//      lanes 16-31 (M=lane-16):half0 -> K c0+8..15, half1 -> K c0+24..31
//    D layout: lane 0 vgpr v -> (M=v, N=0); lane 16 vgpr v -> (M=v+8, N=0).
// ---------------------------------------------------------------------------
__global__ __launch_bounds__(32) void softmax_wmma_kernel(const float* __restrict__ plabel,
                                                          float* __restrict__ probsT) {
  const int box0 = blockIdx.x * 16;
  const int lane = threadIdx.x;
  const int mrow = lane & 15;
  const bool hi  = lane >= 16;
  int n = box0 + mrow;
  if (n >= NBOX) n = NBOX - 1;          // clamp: keep EXEC all-ones for WMMA

  v16h bOnes;
#pragma unroll
  for (int i = 0; i < 16; ++i) bOnes[i] = (_Float16)1.0f;

  v8f acc = {};
#pragma unroll
  for (int chunk = 0; chunk < 3; ++chunk) {
    const int base = chunk * 32 + (hi ? 8 : 0);
    v16h a;
#pragma unroll
    for (int i = 0; i < 16; ++i) {
      int c = base + (i < 8 ? i : i + 8);
      float e = (c < NCLS) ? __expf(plabel[c * NBOX + n]) : 0.0f;
      a[i] = (_Float16)e;
    }
    acc = __builtin_amdgcn_wmma_f32_16x16x32_f16(
        /*neg_a=*/false, a, /*neg_b=*/false, bOnes,
        /*c_mod=*/(short)0, acc, /*reuse_a=*/false, /*reuse_b=*/false);
  }

  __shared__ float denom[16];
  if (lane == 0) {
#pragma unroll
    for (int v = 0; v < 8; ++v) denom[v] = acc[v];        // M = 0..7
  }
  if (lane == 16) {
#pragma unroll
    for (int v = 0; v < 8; ++v) denom[8 + v] = acc[v];    // M = 8..15
  }
  __syncthreads();

  // Write class-major probs for the 80 foreground classes.
  for (int t = lane; t < NFG * 16; t += 32) {
    int b = t & 15;
    int c = (t >> 4) + 1;                 // classes 1..80
    int nb = box0 + b;
    if (nb < NBOX) {
      float e = __expf(plabel[c * NBOX + nb]);
      probsT[(c - 1) * NBOX + nb] = e / denom[b];
    }
  }
}

// ---------------------------------------------------------------------------
// 4) Per-class top-200 + greedy NMS. One workgroup per class; scores live in
//    LDS (35 KB of the 320 KB WGP pool).
// ---------------------------------------------------------------------------
__device__ inline float iou_pair(const float* a, const float* b) {
  float lx = fmaxf(a[0], b[0]), ly = fmaxf(a[1], b[1]);
  float rx = fminf(a[2], b[2]), ry = fminf(a[3], b[3]);
  float w = fmaxf(rx - lx, 0.0f), h = fmaxf(ry - ly, 0.0f);
  float inter = w * h;
  float aa = (a[2] - a[0]) * (a[3] - a[1]);
  float ab = (b[2] - b[0]) * (b[3] - b[1]);
  return inter / (aa + ab - inter);
}

__global__ __launch_bounds__(256) void nms_kernel(const float* __restrict__ probsT,
                                                  const float* __restrict__ ltrb,
                                                  float* __restrict__ scores_c,
                                                  float* __restrict__ boxes_c) {
  const int cls = blockIdx.x;                // 0..79
  const int tid = threadIdx.x;
  __shared__ float s[NBOX];
  __shared__ float rv[256];
  __shared__ int   ri[256];
  __shared__ float tv[TOPK];
  __shared__ int   ti[TOPK];
  __shared__ float cb[TOPK][4];
  __shared__ int   keep[TOPK];
  __shared__ int   sup;

  const float* sc = probsT + (size_t)cls * NBOX;
  for (int i = tid; i < NBOX; i += 256) s[i] = sc[i];
  __syncthreads();

  // top-200 via repeated parallel argmax (ties -> lower index, matches top_k)
  for (int k = 0; k < TOPK; ++k) {
    float bv = -1e30f; int bi = 0x7fffffff;
    for (int i = tid; i < NBOX; i += 256) {
      float v = s[i];
      if (v > bv) { bv = v; bi = i; }
    }
    rv[tid] = bv; ri[tid] = bi;
    __syncthreads();
    for (int off = 128; off > 0; off >>= 1) {
      if (tid < off) {
        bool take = (rv[tid + off] > rv[tid]) ||
                    (rv[tid + off] == rv[tid] && ri[tid + off] < ri[tid]);
        if (take) { rv[tid] = rv[tid + off]; ri[tid] = ri[tid + off]; }
      }
      __syncthreads();
    }
    if (tid == 0) { tv[k] = rv[0]; ti[k] = ri[0]; s[ri[0]] = -1e30f; }
    __syncthreads();
  }

  for (int i = tid; i < TOPK; i += 256) {
    int b = ti[i];
    cb[i][0] = ltrb[b * 4 + 0];
    cb[i][1] = ltrb[b * 4 + 1];
    cb[i][2] = ltrb[b * 4 + 2];
    cb[i][3] = ltrb[b * 4 + 3];
    keep[i] = 0;
  }
  __syncthreads();

  // greedy suppression: sup_i = any_j<i (iou>0.5 && keep_j)
  for (int i = 0; i < TOPK; ++i) {
    if (tid == 0) sup = 0;
    __syncthreads();
    if (tid < i && keep[tid]) {
      if (iou_pair(cb[i], cb[tid]) > 0.5f) atomicOr(&sup, 1);
    }
    __syncthreads();
    if (tid == 0) keep[i] = (tv[i] > 0.05f) && (sup == 0);
    __syncthreads();
  }

  for (int i = tid; i < TOPK; i += 256) {
    int o = cls * TOPK + i;
    scores_c[o] = keep[i] ? tv[i] : 0.0f;
    boxes_c[o * 4 + 0] = cb[i][0];
    boxes_c[o * 4 + 1] = cb[i][1];
    boxes_c[o * 4 + 2] = cb[i][2];
    boxes_c[o * 4 + 3] = cb[i][3];
  }
}

// ---------------------------------------------------------------------------
// 5) Global top-20 + final output assembly.
//    d_out layout: image[270000] | boxes[20*4] | labels[20] | scores[20] | valid[20]
// ---------------------------------------------------------------------------
__global__ __launch_bounds__(256) void final_kernel(const float* __restrict__ scores_c,
                                                    const float* __restrict__ boxes_c,
                                                    float* __restrict__ out) {
  const int tid = threadIdx.x;
  const int N = NFG * TOPK;                  // 16000
  __shared__ float s[NFG * TOPK];
  __shared__ float rv[256];
  __shared__ int   ri[256];
  __shared__ float tv[MAXOUT];
  __shared__ int   ti[MAXOUT];

  for (int i = tid; i < N; i += 256) s[i] = scores_c[i];
  __syncthreads();

  for (int k = 0; k < MAXOUT; ++k) {
    float bv = -1e30f; int bi = 0x7fffffff;
    for (int i = tid; i < N; i += 256) {
      float v = s[i];
      if (v > bv) { bv = v; bi = i; }
    }
    rv[tid] = bv; ri[tid] = bi;
    __syncthreads();
    for (int off = 128; off > 0; off >>= 1) {
      if (tid < off) {
        bool take = (rv[tid + off] > rv[tid]) ||
                    (rv[tid + off] == rv[tid] && ri[tid + off] < ri[tid]);
        if (take) { rv[tid] = rv[tid + off]; ri[tid] = ri[tid + off]; }
      }
      __syncthreads();
    }
    if (tid == 0) { tv[k] = rv[0]; ti[k] = ri[0]; s[ri[0]] = -1e30f; }
    __syncthreads();
  }

  if (tid < MAXOUT) {
    const int OUT_IMG = 3 * OH * OW;        // 270000
    float sc  = tv[tid];
    int   fi  = ti[tid];
    int   cls = fi / TOPK;                  // 0..79 -> label cls+1 -> COCO[cls]
    bool  ok  = sc > 0.2f;
    float* ob = out + OUT_IMG;
#pragma unroll
    for (int j = 0; j < 4; ++j) {
      float b = fminf(fmaxf(boxes_c[fi * 4 + j], 0.0f), 1.0f) * 4096.0f;
      ob[tid * 4 + j] = ok ? b : 0.0f;
    }
    out[OUT_IMG + 80 + tid]  = ok ? (float)c_coco[cls] : 0.0f;  // labels
    out[OUT_IMG + 100 + tid] = ok ? sc : 0.0f;                  // scores
    out[OUT_IMG + 120 + tid] = ok ? 1.0f : 0.0f;                // valid
  }
}

// ---------------------------------------------------------------------------
extern "C" void kernel_launch(void* const* d_in, const int* in_sizes, int n_in,
                              void* d_out, int out_size, void* d_ws, size_t ws_size,
                              hipStream_t stream) {
  const float* image  = (const float*)d_in[0];   // [3,4096,4096]
  const float* ploc   = (const float*)d_in[1];   // [1,4,8732]
  const float* plabel = (const float*)d_in[2];   // [1,81,8732]
  const float* dboxes = (const float*)d_in[3];   // [8732,4]
  float* out = (float*)d_out;

  // Workspace layout (floats)
  float* ws = (float*)d_ws;
  float* ltrb     = ws;                                  // 8732*4   = 34928
  float* probsT   = ltrb + NBOX * 4;                     // 80*8732  = 698560
  float* scores_c = probsT + (size_t)NFG * NBOX;         // 16000
  float* boxes_c  = scores_c + NFG * TOPK;               // 64000

  // 1) fused normalize+resize (gather path: ~4 MB touched instead of 201 MB)
  {
    int n = 3 * OH * OW;
    resize_kernel<<<(n + 255) / 256, 256, 0, stream>>>(image, out);
  }
  // 2) decode boxes
  decode_kernel<<<(NBOX + 255) / 256, 256, 0, stream>>>(ploc, dboxes, ltrb);
  // 3) softmax with WMMA row-reduction (16 boxes per wave32)
  {
    int nwaves = (NBOX + 15) / 16;   // 546
    softmax_wmma_kernel<<<nwaves, 32, 0, stream>>>(plabel, probsT);
  }
  // 4) per-class top-200 + greedy NMS (80 workgroups in parallel)
  nms_kernel<<<NFG, 256, 0, stream>>>(probsT, ltrb, scores_c, boxes_c);
  // 5) global top-20 + output assembly
  final_kernel<<<1, 256, 0, stream>>>(scores_c, boxes_c, out);
}
